// LSTMModel_72181220376611
// MI455X (gfx1250) — matrix operand order repaired and checked by reading, not compiled, not run on hardware
//
#include <hip/hip_runtime.h>

// ---------------------------------------------------------------------------
// Persistent 2-layer LSTM (H=2048, T=2048, FUTURE=128) for gfx1250 (MI455X).
// f16 weights (100.6 MB, L2-resident), GEMV via v_wmma_f32_16x16x32_f16,
// fp32 accumulate + fp32 gate math. 128 WGs x 256 thr (8 waves = 4 gates x
// 2 K-halves, 16 outputs/WG), ONE grid barrier per step in the main phase,
// two in the autoregressive future phase. c1/c2 live in wave-0 registers.
// ---------------------------------------------------------------------------

#define HH    2048
#define TT    2048
#define FUT   128
#define NSTEP (TT + FUT)
#define NWG   128
#define KHALF 1024
#define KIT2  (KHALF / 32)   // 32 K-iterations of K=32 per half-tile

typedef __attribute__((ext_vector_type(8)))  _Float16 v8h;
typedef __attribute__((ext_vector_type(16))) _Float16 v16h;
typedef __attribute__((ext_vector_type(8)))  float    v8f;

__device__ __forceinline__ float sigf(float v) { return 1.0f / (1.0f + __expf(-v)); }
// stable at both ends: e->inf => 1, e->0 => -1
__device__ __forceinline__ float tanf_(float v) { return 1.0f - 2.0f / (__expf(2.0f * v) + 1.0f); }

// Half-K GEMV tile: acc += A(h row-0) x B(W slab), K = 1024.
// wr already offset to (row, ks*KHALF + hi*16); ha already offset by ks*KHALF.
__device__ __forceinline__ v8f mac_tile(const _Float16* __restrict__ wr,
                                        const _Float16* __restrict__ ha,
                                        int hi8, v8f acc)
{
  #pragma unroll 4
  for (int it = 0; it < KIT2; ++it) {
    const int kb = it * 32;
    v16h b = *(const v16h*)(wr + kb);                    // 32B weight slab (2x b128)
    v8h alo = *(const v8h*)(ha + kb + 2 * hi8);          // A 16x32 f16 layout
    v8h ahi = *(const v8h*)(ha + kb + 2 * hi8 + 8);
    v16h a = __builtin_shufflevector(alo, ahi,
                                     0,1,2,3,4,5,6,7,8,9,10,11,12,13,14,15);
    acc = __builtin_amdgcn_wmma_f32_16x16x32_f16(false, a, false, b,
                                                 (short)0, acc, false, false);
  }
  return acc;
}

// Grid barrier: arrival counter at syncv[0], generation at syncv[64] (256B apart).
__device__ __forceinline__ void gbar(int* __restrict__ syncv, int target)
{
  __threadfence();                      // release our global stores
  __syncthreads();
  if (threadIdx.x == 0) {
    if (__hip_atomic_fetch_add(&syncv[0], 1, __ATOMIC_ACQ_REL,
                               __HIP_MEMORY_SCOPE_AGENT) == NWG - 1) {
      __hip_atomic_store(&syncv[0], 0, __ATOMIC_RELAXED, __HIP_MEMORY_SCOPE_AGENT);
      __hip_atomic_store(&syncv[64], target, __ATOMIC_RELEASE, __HIP_MEMORY_SCOPE_AGENT);
    } else {
      while (__hip_atomic_load(&syncv[64], __ATOMIC_ACQUIRE,
                               __HIP_MEMORY_SCOPE_AGENT) < target)
        __builtin_amdgcn_s_sleep(1);
    }
  }
  __syncthreads();
  __threadfence();                      // acquire: drop stale L0 lines
}

__global__ void cvt_f32_f16(const float* __restrict__ src,
                            _Float16* __restrict__ dst, int n)
{
  int i = blockIdx.x * blockDim.x + threadIdx.x;
  int stride = gridDim.x * blockDim.x;
  for (; i < n; i += stride) dst[i] = (_Float16)src[i];
}

__global__ void init_state(_Float16* __restrict__ h1, _Float16* __restrict__ h2,
                           _Float16* __restrict__ zbuf, int* __restrict__ syncv,
                           float* __restrict__ out, const float* __restrict__ b_lin)
{
  int i = blockIdx.x * blockDim.x + threadIdx.x;
  if (i < 2 * HH) { h1[i] = (_Float16)0.0f; h2[i] = (_Float16)0.0f; }
  if (i < HH)     zbuf[i] = (_Float16)0.0f;
  if (i == 0)     { syncv[0] = 0; syncv[64] = 0; }
  if (i < NSTEP)  out[i] = b_lin[0];
}

__global__ void __launch_bounds__(256)
lstm_persistent(const float* __restrict__ input,
                const float* __restrict__ w_ih1,
                const float* __restrict__ b_ih1, const float* __restrict__ b_hh1,
                const float* __restrict__ b_ih2, const float* __restrict__ b_hh2,
                const float* __restrict__ w_lin,
                const _Float16* __restrict__ W1,   // w_hh1 f16 [4H][H]
                const _Float16* __restrict__ W2,   // w_ih2 f16 [4H][H]
                const _Float16* __restrict__ W3,   // w_hh2 f16 [4H][H]
                _Float16* __restrict__ h1,         // [2][H] double buffer
                _Float16* __restrict__ h2,         // [2][H] double buffer
                const _Float16* __restrict__ zbuf, // H zeros (A rows M!=0)
                int* __restrict__ syncv,
                float* __restrict__ out)
{
  const int wg   = blockIdx.x;
  const int w    = threadIdx.x >> 5;     // wave 0..7
  const int lane = threadIdx.x & 31;
  const int g    = w & 3;                // gate 0..3 (i,f,g,o)
  const int ks   = w >> 2;               // K half 0..1
  const int m    = lane & 15;            // B/D column
  const int hi   = lane >> 4;
  const int hi8  = hi << 3;
  const int nb   = wg * 16;              // WG's 16 output rows per gate
  const int kofs = ks * KHALF;

  const size_t rowoff = (size_t)(g * HH + nb + m) * HH + (size_t)kofs + (size_t)(hi << 4);
  const _Float16* w1r = W1 + rowoff;
  const _Float16* w2r = W2 + rowoff;
  const _Float16* w3r = W3 + rowoff;

  // Hoisted per-row constants (used by wave 0; n = nb + m).
  float bs1[4], wx1[4], bs2[4];
  #pragma unroll
  for (int q = 0; q < 4; ++q) {
    bs1[q] = b_ih1[q * HH + nb + m] + b_hh1[q * HH + nb + m];
    wx1[q] = w_ih1[q * HH + nb + m];
    bs2[q] = b_ih2[q * HH + nb + m] + b_hh2[q * HH + nb + m];
  }
  const float wl = w_lin[nb + m];
  float c1r = 0.0f, c2r = 0.0f;          // cell states: WG-private -> registers

  __shared__ float gbuf[2][4][16];       // [K-half][gate][n]

  int bar = 0;
  for (int t = 0; t < NSTEP; ++t) {
    const int cur = t & 1, nxt = cur ^ 1;
    const _Float16* h1c = h1 + cur * HH;
    _Float16*       h1n = h1 + nxt * HH;
    const _Float16* h2c = h2 + cur * HH;
    _Float16*       h2n = h2 + nxt * HH;

    // ---- layer 1: gates1 = W_hh1 * h1_cur -------------------------------
    {
      const _Float16* ha = ((m == 0) ? h1c : zbuf) + kofs;
      v8f acc = {};
      acc = mac_tile(w1r, ha, hi8, acc);
      if (lane < 16) gbuf[ks][g][lane] = acc[0];   // D row 0 = VGPR0, lanes 0-15
    }
    __syncthreads();
    if (w == 0) {                                   // lanes 16-31 duplicate compute
      const float x = (t < TT) ? input[t] : out[t - 1];
      float gi = gbuf[0][0][m] + gbuf[1][0][m] + bs1[0] + x * wx1[0];
      float gf = gbuf[0][1][m] + gbuf[1][1][m] + bs1[1] + x * wx1[1];
      float gg = gbuf[0][2][m] + gbuf[1][2][m] + bs1[2] + x * wx1[2];
      float go = gbuf[0][3][m] + gbuf[1][3][m] + bs1[3] + x * wx1[3];
      c1r = sigf(gf) * c1r + sigf(gi) * tanf_(gg);
      float hn = sigf(go) * tanf_(c1r);
      if (lane < 16) h1n[nb + m] = (_Float16)hn;
    }
    gbar(syncv, ++bar);                             // h1_new visible grid-wide

    // ---- layer 2: gates2 = W_ih2 * h1_new + W_hh2 * h2_cur --------------
    {
      const _Float16* haA = ((m == 0) ? h1n : zbuf) + kofs;
      const _Float16* haB = ((m == 0) ? h2c : zbuf) + kofs;
      v8f acc = {};
      acc = mac_tile(w2r, haA, hi8, acc);
      acc = mac_tile(w3r, haB, hi8, acc);
      if (lane < 16) gbuf[ks][g][lane] = acc[0];
    }
    __syncthreads();
    float pl = 0.0f;
    if (w == 0) {
      float gi = gbuf[0][0][m] + gbuf[1][0][m] + bs2[0];
      float gf = gbuf[0][1][m] + gbuf[1][1][m] + bs2[1];
      float gg = gbuf[0][2][m] + gbuf[1][2][m] + bs2[2];
      float go = gbuf[0][3][m] + gbuf[1][3][m] + bs2[3];
      c2r = sigf(gf) * c2r + sigf(gi) * tanf_(gg);
      float hn = sigf(go) * tanf_(c2r);
      if (lane < 16) h2n[nb + m] = (_Float16)hn;
      pl = (lane < 16) ? wl * hn : 0.0f;            // out[t] partial
      #pragma unroll
      for (int off = 16; off > 0; off >>= 1) pl += __shfl_down(pl, off, 32);
      if (lane == 0) atomicAdd(&out[t], pl);
    }
    // Main phase: next step's barrier-1 already orders h2/out consumption.
    // Future phase: out[t] feeds x of step t+1 *before* its barrier -> sync now.
    if (t >= TT - 1) gbar(syncv, ++bar);
    __syncthreads();                                // LDS gbuf reuse (WG-local)
  }
}

extern "C" void kernel_launch(void* const* d_in, const int* in_sizes, int n_in,
                              void* d_out, int out_size, void* d_ws, size_t ws_size,
                              hipStream_t stream)
{
  (void)in_sizes; (void)n_in; (void)out_size; (void)ws_size;

  const float* input = (const float*)d_in[0];
  const float* w_ih1 = (const float*)d_in[1];
  const float* w_hh1 = (const float*)d_in[2];
  const float* b_ih1 = (const float*)d_in[3];
  const float* b_hh1 = (const float*)d_in[4];
  const float* w_ih2 = (const float*)d_in[5];
  const float* w_hh2 = (const float*)d_in[6];
  const float* b_ih2 = (const float*)d_in[7];
  const float* b_hh2 = (const float*)d_in[8];
  const float* w_lin = (const float*)d_in[9];
  const float* b_lin = (const float*)d_in[10];
  float* out = (float*)d_out;

  const size_t WN = (size_t)4 * HH * HH;            // 16,777,216 elems/matrix
  char* ws = (char*)d_ws;
  _Float16* W1   = (_Float16*)ws;  ws += WN * sizeof(_Float16);   // 32 MiB
  _Float16* W2   = (_Float16*)ws;  ws += WN * sizeof(_Float16);   // 32 MiB
  _Float16* W3   = (_Float16*)ws;  ws += WN * sizeof(_Float16);   // 32 MiB
  _Float16* h1   = (_Float16*)ws;  ws += 2 * HH * sizeof(_Float16);
  _Float16* h2   = (_Float16*)ws;  ws += 2 * HH * sizeof(_Float16);
  _Float16* zbuf = (_Float16*)ws;  ws += HH * sizeof(_Float16);
  int*      syncv = (int*)ws;      ws += 512;       // cnt @0, gen @+256B

  cvt_f32_f16<<<2048, 256, 0, stream>>>(w_hh1, W1, (int)WN);
  cvt_f32_f16<<<2048, 256, 0, stream>>>(w_ih2, W2, (int)WN);
  cvt_f32_f16<<<2048, 256, 0, stream>>>(w_hh2, W3, (int)WN);
  init_state<<<16, 256, 0, stream>>>(h1, h2, zbuf, syncv, out, b_lin);

  lstm_persistent<<<NWG, 256, 0, stream>>>(
      input, w_ih1, b_ih1, b_hh1, b_ih2, b_hh2, w_lin,
      W1, W2, W3, h1, h2, zbuf, syncv, out);
}
